// AutoregressiveTransformer_43757126812192
// MI455X (gfx1250) — matrix operand order repaired and checked by reading, not compiled, not run on hardware
//
#include <hip/hip_runtime.h>
#include <cstddef>

typedef __bf16 bf16_t;
typedef bf16_t v16bf __attribute__((ext_vector_type(16)));
typedef float  v8f   __attribute__((ext_vector_type(8)));
typedef unsigned int u32x4 __attribute__((ext_vector_type(4)));
typedef unsigned int u32x8 __attribute__((ext_vector_type(8)));

#define ZD 1024
#define BSZ 64
#define NH 16
#define HE 64
#define LSTEPS 20

enum { ACT_NONE = 0, ACT_RELU = 1, ACT_GELU = 2 };

__device__ __forceinline__ unsigned short f2bf(float f) {
  unsigned u = __builtin_bit_cast(unsigned, f);
  u += 0x7FFFu + ((u >> 16) & 1u);      // round-to-nearest-even
  return (unsigned short)(u >> 16);
}

// ---------------------------------------------------------------------------
// TDM: async 2D tile load Global -> LDS (rows x 64 fp32, row stride in elems).
// Builds D# group0 (128b) + group1 (256b) in SGPRs, issues tensor_load_to_lds.
// ---------------------------------------------------------------------------
__device__ __forceinline__ void tdm_load_rows64(const float* gsrc, unsigned lds_off,
                                                int rows, unsigned rowStrideElems) {
  unsigned long long ga = (unsigned long long)gsrc;
  u32x4 g0;
  g0[0] = 1u;                                  // count=1, user descriptor
  g0[1] = lds_off;                             // lds_addr
  g0[2] = (unsigned)(ga & 0xFFFFFFFFu);        // global_addr[31:0]
  g0[3] = (unsigned)(ga >> 32) | 0x80000000u;  // global_addr[56:32] | type=2
  u32x8 g1;
  g1[0] = 0x20000u;                            // data_size=2 (4 bytes), mask=0
  g1[1] = 64u << 16;                           // tensor_dim0 = 64 (lo16)
  g1[2] = ((unsigned)rows & 0xFFFFu) << 16;    // tensor_dim1 lo16 = rows
  g1[3] = 64u << 16;                           // tile_dim0 = 64
  g1[4] = (unsigned)rows & 0xFFFFu;            // tile_dim1 = rows, tile_dim2=0
  g1[5] = rowStrideElems;                      // tensor_dim0_stride[31:0]
  g1[6] = 0u;
  g1[7] = 0u;
  asm volatile("tensor_load_to_lds %0, %1" :: "s"(g0), "s"(g1) : "memory");
}

// ---------------------------------------------------------------------------
// Generic GEMM: C[M,N] = act(A[M,K] @ W[K,N] + bias[N])
// fp32 in / fp32 out, bf16 WMMA compute (v_wmma_f32_16x16x32_bf16).
// Block tile 64x128, 128 threads = 4 waves; each wave owns a 16-row strip and
// all 128 columns (1 A-frag reused across 8 WMMAs per K-step).
// Requires: K % 32 == 0, M % 64 == 0 (guarded stores anyway), any N.
// ---------------------------------------------------------------------------
__global__ __launch_bounds__(128)
void gemm_bf16_wmma(const float* __restrict__ A, const float* __restrict__ W,
                    const float* __restrict__ bias, float* __restrict__ C,
                    int M, int N, int K, int act)
{
  __shared__ unsigned short As[64][36];    // [m][k] bf16, padded row (18 dwords)
  __shared__ unsigned short Ws[128][36];   // [n][k] bf16 (transposed), padded
  __shared__ float biasS[128];

  const int tid  = threadIdx.x;
  const int lane = tid & 31;
  const int wave = tid >> 5;
  const int mBase = blockIdx.y * 64;
  const int nBase = blockIdx.x * 128;

  // ---- bias tile -> LDS via CDNA5 async global->LDS load -------------------
  {
    int n = nBase + tid;
    if (n < N) {
      const float* src = bias + n;
      unsigned lds_off = (unsigned)(unsigned long long)(void*)&biasS[tid];
      asm volatile("global_load_async_to_lds_b32 %0, %1, off"
                   :: "v"(lds_off), "v"(src) : "memory");
    } else {
      biasS[tid] = 0.0f;
    }
  }
  asm volatile("s_wait_asynccnt 0x0" ::: "memory");

  v8f acc[8];
  #pragma unroll
  for (int g = 0; g < 8; ++g)
    #pragma unroll
    for (int i = 0; i < 8; ++i) acc[g][i] = 0.0f;

  const bool nFull = (nBase + 128 <= N);

  for (int kt = 0; kt < K; kt += 32) {
    // ---- A tile: 64x32, thread = (row, half-of-k) ----
    {
      int r  = tid >> 1;
      int hk = (tid & 1) * 16;
      const float* src = A + (size_t)(mBase + r) * K + kt + hk;
      const float4* s4 = (const float4*)src;
      #pragma unroll
      for (int q = 0; q < 4; ++q) {
        float4 f = s4[q];
        float t4[4] = {f.x, f.y, f.z, f.w};
        #pragma unroll
        for (int j = 0; j < 4; ++j) As[r][hk + q * 4 + j] = f2bf(t4[j]);
      }
      if (kt + 32 < K) __builtin_prefetch(src + 32, 0, 0);
    }
    // ---- W tile: 32x128 -> LDS transposed [n][k] ----
    {
      int k    = tid >> 2;
      int nOff = (tid & 3) * 32;
      const float* src = W + (size_t)(kt + k) * N + nBase + nOff;
      if (nFull) {
        const float4* s4 = (const float4*)src;
        #pragma unroll
        for (int q = 0; q < 8; ++q) {
          float4 f = s4[q];
          float t4[4] = {f.x, f.y, f.z, f.w};
          #pragma unroll
          for (int j = 0; j < 4; ++j) Ws[nOff + q * 4 + j][k] = f2bf(t4[j]);
        }
      } else {
        #pragma unroll
        for (int j = 0; j < 32; ++j) {
          int n = nBase + nOff + j;
          float f = (n < N) ? src[j] : 0.0f;
          Ws[nOff + j][k] = f2bf(f);
        }
      }
      if (kt + 32 < K) __builtin_prefetch(src + (size_t)32 * N, 0, 0);
    }
    __syncthreads();

    // ---- fragments (per ISA 16-bit A 16x32 / B 32x16 layouts) ----
    const int half = lane >> 4;
    const int ml   = lane & 15;
    union { v16bf v; unsigned u[8]; } af;
    const unsigned* Arow = (const unsigned*)&As[wave * 16 + ml][0];
    #pragma unroll
    for (int v = 0; v < 4; ++v) {
      af.u[v]     = Arow[half * 4 + v];       // K = half*8 + 2v
      af.u[4 + v] = Arow[8 + half * 4 + v];   // K = 16 + half*8 + 2v
    }
    #pragma unroll
    for (int g = 0; g < 8; ++g) {
      union { v16bf v; unsigned u[8]; } bfm;
      const unsigned* Brow = (const unsigned*)&Ws[g * 16 + ml][0];
      #pragma unroll
      for (int v = 0; v < 8; ++v) bfm.u[v] = Brow[(v >> 2) * 8 + half * 4 + (v & 3)]; // K = (v>>2)*16 + half*8 + 2(v&3)
      acc[g] = __builtin_amdgcn_wmma_f32_16x16x32_bf16(
          false, af.v, false, bfm.v, (short)0, acc[g], false, false);
    }
    __syncthreads();
  }

  // ---- epilogue: bias + activation, C/D layout m = w*16 + half*8 + r -------
  const int half = lane >> 4;
  const int nl   = lane & 15;
  #pragma unroll
  for (int g = 0; g < 8; ++g) {
    int n = nBase + g * 16 + nl;
    if (n >= N) continue;
    float bv = biasS[g * 16 + nl];
    #pragma unroll
    for (int r = 0; r < 8; ++r) {
      int m = mBase + wave * 16 + half * 8 + r;
      if (m >= M) continue;
      float x = acc[g][r] + bv;
      if (act == ACT_RELU)      x = fmaxf(x, 0.0f);
      else if (act == ACT_GELU) x = 0.5f * x * (1.0f + erff(x * 0.70710678118f));
      C[(size_t)m * N + n] = x;
    }
  }
}

// ---------------------------------------------------------------------------
// LayerNorm over Z=1024: out[row] = LN(a[row] (+ b[row])) * g + beta
// ---------------------------------------------------------------------------
__global__ __launch_bounds__(256)
void ln_kernel(const float* __restrict__ a, const float* __restrict__ b,
               const float* __restrict__ gam, const float* __restrict__ bet,
               float* __restrict__ out)
{
  __shared__ float rs[256], rq[256];
  const int row = blockIdx.x, tid = threadIdx.x;
  const float* pa = a + (size_t)row * ZD;
  const float* pb = b ? b + (size_t)row * ZD : nullptr;
  float x[4]; float s = 0.f, q = 0.f;
  #pragma unroll
  for (int i = 0; i < 4; ++i) {
    int c = tid + i * 256;
    float v = pa[c] + (pb ? pb[c] : 0.0f);
    x[i] = v; s += v; q += v * v;
  }
  rs[tid] = s; rq[tid] = q; __syncthreads();
  for (int st = 128; st > 0; st >>= 1) {
    if (tid < st) { rs[tid] += rs[tid + st]; rq[tid] += rq[tid + st]; }
    __syncthreads();
  }
  float mu  = rs[0] * (1.0f / ZD);
  float var = rq[0] * (1.0f / ZD) - mu * mu;
  float inv = rsqrtf(var + 1e-5f);
  #pragma unroll
  for (int i = 0; i < 4; ++i) {
    int c = tid + i * 256;
    out[(size_t)row * ZD + c] = (x[i] - mu) * inv * gam[c] + bet[c];
  }
}

// ---------------------------------------------------------------------------
// Attention: softmax(Q K^T / 8) V, head dim 64, t <= 20 keys.
// K/V tiles (t rows x 64, row stride 64*ZD elems) are DMA'd to LDS by the
// Tensor Data Mover; compute runs out of LDS.
// Q/K/V rows are token-major: row = tok*64 + b. grid = (B*H, S), block = 64.
// qTokMajor=0 -> Q/output row = b (cross attention, S==1).
// ---------------------------------------------------------------------------
__global__ __launch_bounds__(64)
void attn_kernel(const float* __restrict__ Q, const float* __restrict__ K,
                 const float* __restrict__ V, float* __restrict__ O,
                 int t, int qTokMajor)
{
  __shared__ float Ks[LSTEPS * HE];
  __shared__ float Vs[LSTEPS * HE];
  __shared__ float sc[32];
  __shared__ float qs[64];
  const int bh = blockIdx.x;
  const int b = bh >> 4, h = bh & 15;
  const int s = blockIdx.y;
  const int e = threadIdx.x;

  // wave 0 issues the two TDM tile loads (EXEC ignored; per-wave issue)
  if (threadIdx.x < 32) {
    const size_t rowOff = (size_t)b * ZD + (size_t)h * HE;
    tdm_load_rows64(K + rowOff, (unsigned)(unsigned long long)(void*)&Ks[0],
                    t, (unsigned)(BSZ * ZD));
    tdm_load_rows64(V + rowOff, (unsigned)(unsigned long long)(void*)&Vs[0],
                    t, (unsigned)(BSZ * ZD));
  }
  __builtin_amdgcn_s_wait_tensorcnt(0);

  const int qrow = qTokMajor ? (s * 64 + b) : b;
  qs[e] = Q[(size_t)qrow * ZD + h * HE + e];
  __syncthreads();

  if (e < t) {
    const float* kr = &Ks[e * HE];
    float d = 0.f;
    #pragma unroll 8
    for (int i = 0; i < HE; ++i) d += qs[i] * kr[i];
    sc[e] = d * 0.125f;
  }
  __syncthreads();

  float mx = -1e30f;
  for (int j = 0; j < t; ++j) mx = fmaxf(mx, sc[j]);
  float den = 0.f;
  for (int j = 0; j < t; ++j) den += __expf(sc[j] - mx);
  float o = 0.f;
  for (int j = 0; j < t; ++j)
    o += __expf(sc[j] - mx) * Vs[j * HE + e];
  o /= den;
  const int orow = qTokMajor ? (s * 64 + b) : b;
  O[(size_t)orow * ZD + h * HE + e] = o;
}

// X[0][b][:] = emb[0][:]
__global__ __launch_bounds__(256)
void init_x0_kernel(float* __restrict__ X, const float* __restrict__ emb)
{
  int b = blockIdx.x, tid = threadIdx.x;
  #pragma unroll
  for (int i = 0; i < 4; ++i) {
    int c = tid + i * 256;
    X[(size_t)b * ZD + c] = emb[c];
  }
}

// box: [l][b][64] token-major -> out [b][l][64]
__global__ __launch_bounds__(256)
void scatter_box_kernel(const float* __restrict__ src, float* __restrict__ dst)
{
  int idx = blockIdx.x * 256 + threadIdx.x;   // l*4096 + b*64 + j, total 81920
  if (idx >= BSZ * LSTEPS * 64) return;
  int j = idx & 63, b = (idx >> 6) & 63, l = idx >> 12;
  dst[(size_t)b * (LSTEPS * 64) + l * 64 + j] = src[(size_t)(l * 64 + b) * 64 + j];
}

// comp: [l][b] -> out [b][l]
__global__ __launch_bounds__(256)
void scatter_comp_kernel(const float* __restrict__ src, float* __restrict__ dst)
{
  int idx = blockIdx.x * 256 + threadIdx.x;   // l*64 + b, total 1280
  if (idx >= BSZ * LSTEPS) return;
  int b = idx & 63, l = idx >> 6;
  dst[(size_t)b * LSTEPS + l] = src[(size_t)l * 64 + b];
}

// ---------------------------------------------------------------------------
extern "C" void kernel_launch(void* const* d_in, const int* in_sizes, int n_in,
                              void* d_out, int out_size, void* d_ws, size_t ws_size,
                              hipStream_t stream)
{
  (void)in_sizes; (void)n_in; (void)out_size; (void)ws_size;
  auto in = [&](int i) { return (const float*)d_in[i]; };

  // workspace layout (floats)
  float* W0 = (float*)d_ws;
  size_t off = 0;
  auto alloc = [&](size_t n) { float* p = W0 + off; off += n; return p; };
  float* X    = alloc((size_t)(LSTEPS + 1) * BSZ * ZD);  // token-major feats
  float* zmid = alloc((size_t)BSZ * 2048);
  float* z    = alloc((size_t)BSZ * ZD);
  float* x1   = alloc((size_t)BSZ * ZD);
  float* qc   = alloc((size_t)BSZ * ZD);
  float* P0   = alloc((size_t)LSTEPS * BSZ * ZD);
  float* P1   = alloc((size_t)LSTEPS * BSZ * ZD);
  float* P2   = alloc((size_t)LSTEPS * BSZ * ZD);
  float* P3   = alloc((size_t)LSTEPS * BSZ * ZD);
  float* P4   = alloc((size_t)LSTEPS * BSZ * ZD);
  float* d2   = alloc((size_t)BSZ * ZD);
  float* d3   = alloc((size_t)BSZ * ZD);
  float* d4   = alloc((size_t)BSZ * ZD);
  float* d5   = alloc((size_t)BSZ * ZD);
  float* d6   = alloc((size_t)BSZ * ZD);
  float* d7   = alloc((size_t)BSZ * ZD);
  float* d8   = alloc((size_t)BSZ * ZD);
  float* bh_  = alloc((size_t)LSTEPS * BSZ * 512);
  float* boxb = alloc((size_t)LSTEPS * BSZ * 64);
  float* chh  = alloc((size_t)LSTEPS * BSZ * 128);
  float* cmpb = alloc((size_t)LSTEPS * BSZ);

  auto gemm = [&](const float* A, const float* Wt, const float* bias,
                  float* C, int M, int N, int K, int act) {
    dim3 g((N + 127) / 128, (M + 63) / 64);
    gemm_bf16_wmma<<<g, 128, 0, stream>>>(A, Wt, bias, C, M, N, K, act);
  };
  auto ln = [&](const float* a, const float* b, int gi, int bi, float* o, int rows) {
    ln_kernel<<<rows, 256, 0, stream>>>(a, b, in(gi), in(bi), o);
  };

  // ---- flatten MLP: [64,64000] -> relu -> 2048 -> 1024 ----
  gemm(in(0), in(2), in(3), zmid, BSZ, 2048, 64000, ACT_RELU);
  gemm(zmid, in(4), in(5), z, BSZ, ZD, 2048, ACT_NONE);

  // ---- start token ----
  init_x0_kernel<<<BSZ, 256, 0, stream>>>(X, in(1));

  // ---- decoder self-attn (T=1 => softmax==1) + ln1, cross-Q: step-invariant
  gemm(z, in(28), in(29), d2, BSZ, ZD, ZD, ACT_NONE);   // v = z wv_s + bv_s
  gemm(d2, in(30), in(31), d3, BSZ, ZD, ZD, ACT_NONE);  // self_o = v wo_s + bo_s
  ln(z, d3, 44, 45, x1, BSZ);                           // x1 = ln1(z + self_o)
  gemm(x1, in(32), in(33), qc, BSZ, ZD, ZD, ACT_NONE);  // qc = x1 wq_c + bq_c

  for (int idx = 0; idx < LSTEPS; ++idx) {
    const int t = idx + 1;
    const int M = t * BSZ;
    // -------- encoder over X[0..t) --------
    gemm(X, in(6), in(7), P0, M, ZD, ZD, ACT_NONE);     // Q
    gemm(X, in(8), in(9), P1, M, ZD, ZD, ACT_NONE);     // K
    gemm(X, in(10), in(11), P2, M, ZD, ZD, ACT_NONE);   // V
    attn_kernel<<<dim3(BSZ * NH, t), 64, 0, stream>>>(P0, P1, P2, P3, t, 1);
    gemm(P3, in(12), in(13), P0, M, ZD, ZD, ACT_NONE);  // O = att wo + bo
    ln(X, P0, 18, 19, P1, M);                           // Xa = ln1(X + O)
    gemm(P1, in(14), in(15), P2, M, ZD, ZD, ACT_GELU);  // F1 = gelu(Xa w1 + b1)
    gemm(P2, in(16), in(17), P0, M, ZD, ZD, ACT_NONE);  // F2
    ln(P1, P0, 20, 21, P3, M);                          // Xb = ln2(Xa + F2)
    ln(P3, nullptr, 22, 23, P4, M);                     // Xe = ln_f(Xb)
    // -------- decoder cross + ffn --------
    gemm(P4, in(34), in(35), P0, M, ZD, ZD, ACT_NONE);  // Kc
    gemm(P4, in(36), in(37), P1, M, ZD, ZD, ACT_NONE);  // Vc
    attn_kernel<<<dim3(BSZ * NH, 1), 64, 0, stream>>>(qc, P0, P1, d2, t, 0);
    gemm(d2, in(38), in(39), d3, BSZ, ZD, ZD, ACT_NONE);// Oc
    ln(x1, d3, 46, 47, d4, BSZ);                        // x2 = ln2(x1 + Oc)
    gemm(d4, in(40), in(41), d5, BSZ, ZD, ZD, ACT_GELU);// F1d
    gemm(d5, in(42), in(43), d6, BSZ, ZD, ZD, ACT_NONE);// F2d
    ln(d4, d6, 48, 49, d7, BSZ);                        // x3 = ln3(x2 + F2d)
    ln(d7, nullptr, 50, 51, d8, BSZ);                   // xf = ln_f(x3)
    // -------- per-step MLP -> new token --------
    const float* sw1 = in(52) + (size_t)idx * ZD * ZD;
    const float* sb1 = in(53) + (size_t)idx * ZD;
    const float* sw2 = in(54) + (size_t)idx * ZD * ZD;
    const float* sb2 = in(55) + (size_t)idx * ZD;
    gemm(d8, sw1, sb1, d2, BSZ, ZD, ZD, ACT_RELU);      // h1
    gemm(d2, sw2, sb2, X + (size_t)t * BSZ * ZD, BSZ, ZD, ZD, ACT_RELU); // X[t]
  }

  // ---- heads over obj = X[1..20] (token-major, M = 1280) ----
  const float* obj = X + (size_t)BSZ * ZD;
  const int Mo = LSTEPS * BSZ;
  gemm(obj, in(56), in(57), bh_, Mo, 512, ZD, ACT_RELU);
  gemm(bh_, in(58), in(59), boxb, Mo, 64, 512, ACT_NONE);
  gemm(obj, in(60), in(61), chh, Mo, 128, ZD, ACT_RELU);
  gemm(chh, in(62), in(63), cmpb, Mo, 1, 128, ACT_NONE);

  float* out = (float*)d_out;
  scatter_box_kernel<<<(BSZ * LSTEPS * 64 + 255) / 256, 256, 0, stream>>>(boxb, out);
  scatter_comp_kernel<<<(BSZ * LSTEPS + 255) / 256, 256, 0, stream>>>(
      cmpb, out + (size_t)BSZ * LSTEPS * 64);
}
